// Multi_head_inter_attention_51651276702564
// MI455X (gfx1250) — matrix operand order repaired and checked by reading, not compiled, not run on hardware
//
#include <hip/hip_runtime.h>

typedef __attribute__((ext_vector_type(16))) _Float16 v16h;
typedef __attribute__((ext_vector_type(8)))  float    v8f;

#define NN 207
#define DD 64
#define HH 8
#define TT 12
#define BB 512
#define BN (BB * NN)        // 105984
#define TILES (BN / 16)     // 6624 (exact)
#define TPW 2               // tiles per wave
#define NBLK (TILES / TPW)  // 3312

struct alignas(4) H2 { _Float16 x, y; };

// Compiler scheduling fence: stops LICM/GVN from hoisting loads across
// loop iterations into huge live ranges (which the allocator then spills).
#define SCHED_FENCE() asm volatile("" ::: "memory")

__device__ __forceinline__ v8f wmma_f16(v16h a, v16h b, v8f c) {
  // D = A(16x32,f16) * B(32x16,f16) + C(16x16,f32)
  return __builtin_amdgcn_wmma_f32_16x16x32_f16(false, a, false, b, (short)0, c,
                                                false, false);
}

// A fragment (16x32 f16) from a row-major 16x64 LDS tile, K-block kb (0 or 32).
// lane L row M=L&15 holds K = kb+j for lanes 0..15, K = kb+16+j for lanes 16..31.
__device__ __forceinline__ v16h a_frag_f16(const _Float16* src, int kb, int lane) {
  const int r = lane & 15;
  const int hi16 = (lane >> 4) << 4;
  v16h a;
#pragma unroll
  for (int j = 0; j < 16; ++j) {
    int K = kb + j + hi16;
    a[j] = src[r * 64 + K];
  }
  return a;
}

__device__ __forceinline__ v16h a_frag_f32(const float* src, int kb, int lane) {
  const int r = lane & 15;
  const int hi16 = (lane >> 4) << 4;
  v16h a;
#pragma unroll
  for (int j = 0; j < 16; ++j) {
    int K = kb + j + hi16;
    a[j] = (_Float16)src[r * 64 + K];
  }
  return a;
}

// B fragment (32x16 f16) from a row-major 64x64 f32 weight in global memory.
// lanes 0..15 hold K = kb+j, lanes 16..31 hold K = kb+16+j; col = ct*16+(lane&15).
__device__ __forceinline__ v16h b_frag_w(const float* __restrict__ W, int kb, int ct,
                                         int lane) {
  const int col = ct * 16 + (lane & 15);
  const int hi16 = (lane >> 4) << 4;
  v16h b;
#pragma unroll
  for (int j = 0; j < 16; ++j) {
    int K = kb + j + hi16;
    b[j] = (_Float16)W[K * 64 + col];
  }
  return b;
}

__global__ __launch_bounds__(32)
__attribute__((amdgpu_num_vgpr(256))) void mhia_kernel(
    const float* __restrict__ enc, const float* __restrict__ dec,
    const int* __restrict__ tp,
    const float* __restrict__ Wq, const float* __restrict__ bq,
    const float* __restrict__ Wk, const float* __restrict__ bk,
    const float* __restrict__ Wv, const float* __restrict__ bv,
    const float* __restrict__ W1, const float* __restrict__ b1,
    float* __restrict__ out, float* __restrict__ att) {
  __shared__ alignas(16) _Float16 s_kv[TT * 16 * 64];  // kv in (+pe) -> V, f16: 24 KB
  __shared__ alignas(32) _Float16 s_bf[16 * 32 * 16];  // per-lane Wk/Wv frags: 16 KB
  __shared__ alignas(16) float s_q[16 * 64];   // q staged -> Q -> ctx: 4 KB
  __shared__ alignas(16) float s_k[16 * 64];   // per-t projected K: 4 KB
  __shared__ alignas(16) float s_peE[TT * 64]; // pe_enc: 3 KB
  __shared__ alignas(16) float s_peD[64];      // pe_dec[t]

  const int lane = threadIdx.x;
  const int colL = lane & 15;
  const int hiL = lane >> 4;
  const int t = tp[0];

  // ---- sinusoidal positional encodings (once per wave) ----
  for (int idx = lane; idx < TT * 64; idx += 32) {
    int pos = idx >> 6, d = idx & 63;
    float expo = (float)(d >> 1) * (1.0f / 32.0f);       // 2*floor(d/2)/64
    float ang = (float)pos * __powf(10000.0f, -expo);
    s_peE[idx] = (d & 1) ? __cosf(ang) : __sinf(ang);
  }
  for (int d = lane; d < 64; d += 32) {
    float expo = (float)(d >> 1) * (1.0f / 32.0f);
    float ang = (float)t * __powf(10000.0f, -expo);
    s_peD[d] = (d & 1) ? __cosf(ang) : __sinf(ang);
  }

  // ---- build Wk/Wv fragments once, park per-lane slices in LDS (16 KB) ----
  // slot(fi) = s_bf + (fi*32 + lane)*16 : 32B aligned, lane-private.
#pragma unroll
  for (int ct = 0; ct < 4; ++ct) {
    *(v16h*)(s_bf + ((2 * ct) * 32 + lane) * 16) = b_frag_w(Wk, 0, ct, lane);
    *(v16h*)(s_bf + ((2 * ct + 1) * 32 + lane) * 16) = b_frag_w(Wk, 32, ct, lane);
    *(v16h*)(s_bf + ((8 + 2 * ct) * 32 + lane) * 16) = b_frag_w(Wv, 0, ct, lane);
    *(v16h*)(s_bf + ((9 + 2 * ct) * 32 + lane) * 16) = b_frag_w(Wv, 32, ct, lane);
  }

  // ---- per-lane bias values for this lane's 4 output column groups ----
  float bkr[4], bvr[4];
#pragma unroll
  for (int ct = 0; ct < 4; ++ct) {
    bkr[ct] = bk[ct * 16 + colL];
    bvr[ct] = bv[ct * 16 + colL];
  }
  __syncthreads();

  // lane -> 4 (pair, head) assignments for the attention phase
  int pi[4], hh[4];
#pragma unroll
  for (int i = 0; i < 4; ++i) {
    int ph = lane + 32 * i;  // 0..127 = 16 pairs x 8 heads
    pi[i] = ph >> 3;
    hh[i] = ph & 7;
  }

  for (int it2 = 0; it2 < TPW; ++it2) {
    const int tile = blockIdx.x * TPW + it2;
    SCHED_FENCE();    // keep per-tile loads (incl. Wq/W1 gathers) inside the tile
    __syncthreads();  // previous tile fully done with LDS

    // ---- stage q rows, vectorized (dec row g is contiguous at g*64) ----
    {
      const float4* dsrc = (const float4*)(dec + (long)tile * 1024);
      const float4* pe4 = (const float4*)s_peD;
      float4* qdst = (float4*)s_q;
#pragma unroll
      for (int i4 = 0; i4 < 8; ++i4) {
        int idx = lane + 32 * i4;  // 256 float4s
        float4 v = dsrc[idx];
        float4 p = pe4[idx & 15];
        qdst[idx] = make_float4(v.x + p.x, v.y + p.y, v.z + p.z, v.w + p.w);
      }
    }

    // ---- stage kv tile (+pe_enc) as f16, float2 per lane per row ----
    for (int r = 0; r < 16; ++r) {
      int g = tile * 16 + r;
      int b = g / NN;
      int n = g - b * NN;
      const float* rowb = enc + ((long)b * TT) * (NN * 64) + (long)n * 64;
      for (int tt = 0; tt < TT; ++tt) {
        float2 v = ((const float2*)(rowb + (long)tt * (NN * 64)))[lane];
        float2 p = ((const float2*)(s_peE + tt * 64))[lane];
        H2 h;
        h.x = (_Float16)(v.x + p.x);
        h.y = (_Float16)(v.y + p.y);
        ((H2*)(s_kv + tt * 1024 + r * 64))[lane] = h;
      }
    }
    __syncthreads();

    // ---- Q projection: Q = (q+pe) @ Wq + bq, in place in s_q ----
    {
      v16h a0 = a_frag_f32(s_q, 0, lane);
      v16h a1 = a_frag_f32(s_q, 32, lane);
      __syncthreads();  // all lanes read staged q before overwrite
      SCHED_FENCE();    // keep Wq gathers here, not hoisted out of the tile loop
#pragma unroll
      for (int ct = 0; ct < 4; ++ct) {
        v16h b0 = b_frag_w(Wq, 0, ct, lane);
        v16h b1f = b_frag_w(Wq, 32, ct, lane);
        v8f c = {};
        c = wmma_f16(a0, b0, c);
        c = wmma_f16(a1, b1f, c);
        float bias = bq[ct * 16 + colL];
#pragma unroll
        for (int rr = 0; rr < 8; ++rr)
          s_q[(rr + hiL * 8) * 64 + ct * 16 + colL] = c[rr] + bias;
      }
    }
    __syncthreads();

    // ---- this lane's q values (tt-invariant): keep in registers ----
    float qv[4][8];
#pragma unroll
    for (int i = 0; i < 4; ++i) {
      int base = pi[i] * 64 + hh[i] * 8;
#pragma unroll
      for (int d = 0; d < 8; ++d) qv[i][d] = s_q[base + d];
    }

    // ---- fused K+V projection per time step; scores from K; V f16 in place ----
    float sc[4][TT];
#pragma unroll
    for (int tt = 0; tt < TT; ++tt) {
      SCHED_FENCE();  // keep this iteration's LDS loads inside the iteration
      v16h a0 = a_frag_f16(s_kv + tt * 1024, 0, lane);
      v16h a1 = a_frag_f16(s_kv + tt * 1024, 32, lane);
      // K_tt -> s_k (f32)
#pragma unroll
      for (int ct = 0; ct < 4; ++ct) {
        v16h b0 = *(const v16h*)(s_bf + ((2 * ct) * 32 + lane) * 16);
        v16h b1f = *(const v16h*)(s_bf + ((2 * ct + 1) * 32 + lane) * 16);
        v8f c = {};
        c = wmma_f16(a0, b0, c);
        c = wmma_f16(a1, b1f, c);
#pragma unroll
        for (int rr = 0; rr < 8; ++rr)
          s_k[(rr + hiL * 8) * 64 + ct * 16 + colL] = c[rr] + bkr[ct];
      }
      // V_tt -> overwrite s_kv[tt] (f16); input rows already consumed into a0/a1
#pragma unroll
      for (int ct = 0; ct < 4; ++ct) {
        v16h b0 = *(const v16h*)(s_bf + ((8 + 2 * ct) * 32 + lane) * 16);
        v16h b1f = *(const v16h*)(s_bf + ((9 + 2 * ct) * 32 + lane) * 16);
        v8f c = {};
        c = wmma_f16(a0, b0, c);
        c = wmma_f16(a1, b1f, c);
#pragma unroll
        for (int rr = 0; rr < 8; ++rr)
          s_kv[tt * 1024 + (rr + hiL * 8) * 64 + ct * 16 + colL] =
              (_Float16)(c[rr] + bvr[ct]);
      }
      __syncthreads();
#pragma unroll
      for (int i = 0; i < 4; ++i) {
        int base = pi[i] * 64 + hh[i] * 8;
        float s = 0.f;
#pragma unroll
        for (int d = 0; d < 8; ++d) s += qv[i][d] * s_k[base + d];
        sc[i][tt] = s * 0.35355339059f;  // 1/sqrt(DH)
      }
      __syncthreads();
    }

    // ---- softmax (in registers) + att output (float4 stores) ----
#pragma unroll
    for (int i = 0; i < 4; ++i) {
      float m = sc[i][0];
#pragma unroll
      for (int tt = 1; tt < TT; ++tt) m = fmaxf(m, sc[i][tt]);
      float sum = 0.f;
#pragma unroll
      for (int tt = 0; tt < TT; ++tt) {
        float e = __expf(sc[i][tt] - m);
        sc[i][tt] = e;
        sum += e;
      }
      float inv = 1.0f / sum;
#pragma unroll
      for (int tt = 0; tt < TT; ++tt) sc[i][tt] *= inv;
      long g = (long)tile * 16 + pi[i];
      // att[(h*B*N + g), 0, :], 12 floats, 48B-aligned base
      float4* ap = (float4*)(att + ((long)hh[i] * BN + g) * TT);
#pragma unroll
      for (int q4 = 0; q4 < 3; ++q4)
        ap[q4] = make_float4(sc[i][4 * q4], sc[i][4 * q4 + 1], sc[i][4 * q4 + 2],
                             sc[i][4 * q4 + 3]);
    }

    // ---- ctx accumulation from f16 V in s_kv (stable; no barriers needed) ----
    float ctx[4][8];
#pragma unroll
    for (int i = 0; i < 4; ++i)
#pragma unroll
      for (int d = 0; d < 8; ++d) ctx[i][d] = 0.f;

#pragma unroll
    for (int i = 0; i < 4; ++i) {
      int base0 = pi[i] * 64 + hh[i] * 8;
#pragma unroll
      for (int tt = 0; tt < TT; ++tt) {
        float w = sc[i][tt];
        const H2* vp = (const H2*)(s_kv + tt * 1024 + base0);
#pragma unroll
        for (int dp = 0; dp < 4; ++dp) {
          H2 hv = vp[dp];
          ctx[i][2 * dp] += w * (float)hv.x;
          ctx[i][2 * dp + 1] += w * (float)hv.y;
        }
      }
    }

    // ---- ctx -> s_q (head-concat layout matches reference reshuffle) ----
    __syncthreads();
#pragma unroll
    for (int i = 0; i < 4; ++i) {
      int base = pi[i] * 64 + hh[i] * 8;
#pragma unroll
      for (int d = 0; d < 8; ++d) s_q[base + d] = ctx[i][d];
    }
    __syncthreads();

    // ---- output projection: out = ctx @ W1 + b1 ----
    {
      SCHED_FENCE();  // keep W1 gathers here, not hoisted out of the tile loop
      v16h a0 = a_frag_f32(s_q, 0, lane);
      v16h a1 = a_frag_f32(s_q, 32, lane);
#pragma unroll
      for (int ct = 0; ct < 4; ++ct) {
        v16h b0 = b_frag_w(W1, 0, ct, lane);
        v16h b1f = b_frag_w(W1, 32, ct, lane);
        v8f c = {};
        c = wmma_f16(a0, b0, c);
        c = wmma_f16(a1, b1f, c);
        float bias = b1[ct * 16 + colL];
#pragma unroll
        for (int rr = 0; rr < 8; ++rr) {
          long g = (long)tile * 16 + rr + hiL * 8;
          out[g * 64 + ct * 16 + colL] = c[rr] + bias;
        }
      }
    }
  }
}

extern "C" void kernel_launch(void* const* d_in, const int* in_sizes, int n_in,
                              void* d_out, int out_size, void* d_ws, size_t ws_size,
                              hipStream_t stream) {
  (void)in_sizes; (void)n_in; (void)out_size; (void)d_ws; (void)ws_size;
  const float* enc = (const float*)d_in[0];
  const float* dec = (const float*)d_in[1];
  const int* tp = (const int*)d_in[2];
  const float* Wq = (const float*)d_in[3];
  const float* bq = (const float*)d_in[4];
  const float* Wk = (const float*)d_in[5];
  const float* bk = (const float*)d_in[6];
  const float* Wv = (const float*)d_in[7];
  const float* bv = (const float*)d_in[8];
  const float* W1 = (const float*)d_in[9];
  const float* b1 = (const float*)d_in[10];
  float* out = (float*)d_out;
  float* att = out + (long)BB * NN * DD;  // tuple: (out, att) concatenated
  mhia_kernel<<<NBLK, 32, 0, stream>>>(enc, dec, tp, Wq, bq, Wk, bk, Wv, bv, W1,
                                       b1, out, att);
}
